// Model_19670950216045
// MI455X (gfx1250) — matrix operand order repaired and checked by reading, not compiled, not run on hardware
//
#include <hip/hip_runtime.h>
#include <hip/hip_bf16.h>

// ---------------------------------------------------------------------------
// Model constants (from reference)
// ---------------------------------------------------------------------------
#define T_STEPS   8192
#define TOTAL_ROWS 1048576
#define D_SELF    128
#define D_ALLY    64
#define HID       20
#define NOUT      16
#define LOG_STD_BOUND 2.0f

typedef __attribute__((ext_vector_type(16))) _Float16 v16h;
typedef __attribute__((ext_vector_type(8)))  float    v8f;

// Fast activations: v_exp_f32 + v_rcp_f32 (no IEEE divide expansion).
// tanh(x) = 1 - 2/(e^{2x}+1)  -> correct limits at +/-inf, no NaN.
__device__ __forceinline__ float fast_tanh(float x) {
    return fmaf(-2.0f, __builtin_amdgcn_rcpf(__expf(2.0f * x) + 1.0f), 1.0f);
}
__device__ __forceinline__ float fast_sigmoid(float x) {
    return __builtin_amdgcn_rcpf(1.0f + __expf(-x));
}

// ---------------------------------------------------------------------------
// WMMA tile machinery: C[16 x 32] += A[16 x KT] * B[KT x 32]
// A rows = activation rows (f32 in memory, cvt to f16)
// B      = W^T padded N 20->32, prebuilt in registers per wave
// Layouts per CDNA5 ISA 7.12.2:
//   A (16x32 f16): lane L: M = L&15 ; halves j: K = 8*(L>>4) + (j&7) + 16*(j>>3)
//   B (32x16 f16): lane L: N = L&15 ; halves j: K = 8*(L>>4) + (j&7) + 16*(j>>3)
//   C (16x16 f32): lane L: N = L&15 ; vgpr r: M = r + 8*(L>>4)
// ---------------------------------------------------------------------------
template <int KT>
struct BFrag { v16h b[KT / 32][2]; };

template <int KT>
__device__ __forceinline__ void load_bfrag(const float* __restrict__ W /*[HID,KT]*/,
                                           BFrag<KT>& B, int lane) {
    const int lm = lane & 15;
    const int hi = lane >> 4;
#pragma unroll
    for (int s = 0; s < KT / 32; ++s) {
#pragma unroll
        for (int nt = 0; nt < 2; ++nt) {
            const int n = lm + 16 * nt;
            v16h f;
            if (n < HID) {
                const float* w = W + (size_t)n * KT + 32 * s + 8 * hi;
                float tmp[16];
                *(float4*)(tmp + 0)  = ((const float4*)w)[0];
                *(float4*)(tmp + 4)  = ((const float4*)w)[1];
                *(float4*)(tmp + 8)  = ((const float4*)(w + 16))[0];
                *(float4*)(tmp + 12) = ((const float4*)(w + 16))[1];
#pragma unroll
                for (int j = 0; j < 16; ++j) f[j] = (_Float16)tmp[j];
            } else {
#pragma unroll
                for (int j = 0; j < 16; ++j) f[j] = (_Float16)0.0f;
            }
            B.b[s][nt] = f;
        }
    }
}

template <int KT>
__device__ __forceinline__ void gemm_tile(const float* __restrict__ rowp,
                                          const BFrag<KT>& B,
                                          v8f& c0, v8f& c1, int lane) {
    const int hi = lane >> 4;
#pragma unroll
    for (int s = 0; s < KT / 32; ++s) {
        const float* p = rowp + 32 * s + 8 * hi;
        float tmp[16];
        *(float4*)(tmp + 0)  = ((const float4*)p)[0];
        *(float4*)(tmp + 4)  = ((const float4*)p)[1];
        *(float4*)(tmp + 8)  = ((const float4*)(p + 16))[0];
        *(float4*)(tmp + 12) = ((const float4*)(p + 16))[1];
        v16h a;
#pragma unroll
        for (int j = 0; j < 16; ++j) a[j] = (_Float16)tmp[j];
        c0 = __builtin_amdgcn_wmma_f32_16x16x32_f16(false, a, false, B.b[s][0],
                                                    (short)0, c0, false, false);
        c1 = __builtin_amdgcn_wmma_f32_16x16x32_f16(false, a, false, B.b[s][1],
                                                    (short)0, c1, false, false);
    }
}

// ---------------------------------------------------------------------------
// Kernel 0: zero the reduction workspace (sums + counts)
// ---------------------------------------------------------------------------
__global__ void zero_kernel(float* __restrict__ p, int n) {
    int i = blockIdx.x * blockDim.x + threadIdx.x;
    if (i < n) p[i] = 0.0f;
}

// ---------------------------------------------------------------------------
// Kernel 1: ally MLP (WMMA) + fused sorted-segment reduction via run-compressed
// atomics.  One wave handles a 16-row tile.
// ---------------------------------------------------------------------------
__global__ void ally_kernel(const float* __restrict__ ally,
                            const int*   __restrict__ seg,
                            const float* __restrict__ W_ally,
                            const float* __restrict__ b_ally,
                            float* __restrict__ sums,   // [T,HID]
                            float* __restrict__ cnt,    // [T]
                            int nTiles) {
    const int lane   = threadIdx.x & 31;
    const int lm     = lane & 15;
    const int hi     = lane >> 4;
    const int wavesPerBlock = blockDim.x >> 5;
    const int wave   = blockIdx.x * wavesPerBlock + (threadIdx.x >> 5);
    const int nWaves = gridDim.x * wavesPerBlock;

    BFrag<D_ALLY> B;
    load_bfrag<D_ALLY>(W_ally, B, lane);

    const float bias0 = b_ally[lm];                               // cols 0..15
    const float bias1 = (lm < HID - 16) ? b_ally[lm + 16] : 0.0f; // cols 16..19

    for (int tile = wave; tile < nTiles; tile += nWaves) {
        const float* rowp = ally + (size_t)(tile * 16 + lm) * D_ALLY;

        // prefetch this wave's next tile row (gfx1250 global_prefetch_b8)
        if (tile + nWaves < nTiles)
            __builtin_prefetch(ally + (size_t)((tile + nWaves) * 16 + lm) * D_ALLY, 0, 0);

        v8f c0 = {};
        v8f c1 = {};
        gemm_tile<D_ALLY>(rowp, B, c0, c1, lane);

        float v0[8], v1[8];
#pragma unroll
        for (int r = 0; r < 8; ++r) {
            v0[r] = fast_tanh(c0[r] + bias0);
            v1[r] = fast_tanh(c1[r] + bias1);
        }

        // run-compressed segment reduction over 8 consecutive rows
        const int rbase = tile * 16 + 8 * hi;
        int   cur = seg[rbase];
        float a0 = 0.0f, a1 = 0.0f, ac = 0.0f;
#pragma unroll
        for (int r = 0; r < 8; ++r) {
            const int sg = seg[rbase + r];
            if (sg != cur) {
                atomicAdd(&sums[(size_t)cur * HID + lm], a0);
                if (lm < HID - 16) atomicAdd(&sums[(size_t)cur * HID + 16 + lm], a1);
                if (lm == 0) atomicAdd(&cnt[cur], ac);
                a0 = a1 = ac = 0.0f;
                cur = sg;
            }
            a0 += v0[r];
            a1 += v1[r];
            ac += 1.0f;
        }
        atomicAdd(&sums[(size_t)cur * HID + lm], a0);
        if (lm < HID - 16) atomicAdd(&sums[(size_t)cur * HID + 16 + lm], a1);
        if (lm == 0) atomicAdd(&cnt[cur], ac);
    }
}

// ---------------------------------------------------------------------------
// Kernel 2: self MLP (WMMA), stores tanh(self @ W_self^T + b) -> sbuf [T,HID]
// ---------------------------------------------------------------------------
__global__ void self_kernel(const float* __restrict__ self_in,
                            const float* __restrict__ W_self,
                            const float* __restrict__ b_self,
                            float* __restrict__ sbuf,
                            int nTiles) {
    const int lane   = threadIdx.x & 31;
    const int lm     = lane & 15;
    const int hi     = lane >> 4;
    const int wavesPerBlock = blockDim.x >> 5;
    const int wave   = blockIdx.x * wavesPerBlock + (threadIdx.x >> 5);
    const int nWaves = gridDim.x * wavesPerBlock;

    BFrag<D_SELF> B;
    load_bfrag<D_SELF>(W_self, B, lane);

    const float bias0 = b_self[lm];
    const float bias1 = (lm < HID - 16) ? b_self[lm + 16] : 0.0f;

    for (int tile = wave; tile < nTiles; tile += nWaves) {
        const float* rowp = self_in + (size_t)(tile * 16 + lm) * D_SELF;
        v8f c0 = {};
        v8f c1 = {};
        gemm_tile<D_SELF>(rowp, B, c0, c1, lane);

#pragma unroll
        for (int r = 0; r < 8; ++r) {
            const int row = tile * 16 + 8 * hi + r;
            sbuf[(size_t)row * HID + lm] = fast_tanh(c0[r] + bias0);
            if (lm < HID - 16)
                sbuf[(size_t)row * HID + 16 + lm] = fast_tanh(c1[r] + bias1);
        }
    }
}

// ---------------------------------------------------------------------------
// Kernel 3: per-timestep middle stage: avg, cat, xg (LSTM gate inputs),
// value head.  One thread per timestep; weights are lane-uniform (scalar path).
// ---------------------------------------------------------------------------
__global__ void mid_kernel(const float* __restrict__ sbuf,
                           const float* __restrict__ sums,
                           const float* __restrict__ cnt,
                           const float* __restrict__ W_cat, const float* __restrict__ b_cat,
                           const float* __restrict__ W_ih,  const float* __restrict__ b_ih,
                           const float* __restrict__ b_hh,
                           const float* __restrict__ W_vfc, const float* __restrict__ b_vfc,
                           const float* __restrict__ W_v,   const float* __restrict__ b_v,
                           float* __restrict__ xg,          // [T, 4*HID]
                           float* __restrict__ vout) {      // [T]
    const int t = blockIdx.x * blockDim.x + threadIdx.x;
    if (t >= T_STEPS) return;

    float catin[2 * HID];
    const float icnt = __builtin_amdgcn_rcpf(cnt[t]);
#pragma unroll
    for (int j = 0; j < HID; ++j) catin[j] = sbuf[(size_t)t * HID + j];
#pragma unroll
    for (int j = 0; j < HID; ++j) catin[HID + j] = sums[(size_t)t * HID + j] * icnt;

    float co[HID];
#pragma unroll 4
    for (int i = 0; i < HID; ++i) {
        float acc = b_cat[i];
        const float* w = W_cat + i * 2 * HID;
#pragma unroll
        for (int j = 0; j < 2 * HID; ++j) acc = fmaf(w[j], catin[j], acc);
        co[i] = fast_tanh(acc);
    }

#pragma unroll 4
    for (int g = 0; g < 4 * HID; ++g) {
        float acc = b_ih[g] + b_hh[g];
        const float* w = W_ih + g * HID;
#pragma unroll
        for (int j = 0; j < HID; ++j) acc = fmaf(w[j], co[j], acc);
        xg[(size_t)t * 4 * HID + g] = acc;
    }

    float vacc = b_v[0];
#pragma unroll 4
    for (int i = 0; i < HID; ++i) {
        float acc = b_vfc[i];
        const float* w = W_vfc + i * HID;
#pragma unroll
        for (int j = 0; j < HID; ++j) acc = fmaf(w[j], co[j], acc);
        vacc = fmaf(W_v[i], fast_tanh(acc), vacc);
    }
    vout[t] = vacc;
}

// ---------------------------------------------------------------------------
// Kernel 4: sequential LSTM over 8192 steps.  Single wave32: lane k owns
// hidden unit k (k<20); all 80 W_hh rows pinned in VGPRs; h broadcast each
// step with v_readlane_b32 (scalar-operand FMAs); gate inputs prefetched.
// ---------------------------------------------------------------------------
__global__ void __launch_bounds__(32)
lstm_kernel(const float* __restrict__ xg,
            const float* __restrict__ W_hh,
            const float* __restrict__ h0,
            const float* __restrict__ c0,
            float* __restrict__ lstm_buf) {   // [T, HID]
    const int k = threadIdx.x;
    const bool act = (k < HID);

    float Wi[HID], Wf[HID], Wg[HID], Wo[HID];
    if (act) {
#pragma unroll
        for (int j = 0; j < HID; ++j) {
            Wi[j] = W_hh[(0 * HID + k) * HID + j];
            Wf[j] = W_hh[(1 * HID + k) * HID + j];
            Wg[j] = W_hh[(2 * HID + k) * HID + j];
            Wo[j] = W_hh[(3 * HID + k) * HID + j];
        }
    } else {
#pragma unroll
        for (int j = 0; j < HID; ++j) { Wi[j] = Wf[j] = Wg[j] = Wo[j] = 0.0f; }
    }

    float h = act ? h0[k] : 0.0f;
    float c = act ? c0[k] : 0.0f;

    // preload step 0 gate inputs
    float gi = 0.f, gf = 0.f, gg = 0.f, go = 0.f;
    if (act) {
        const float* p = xg;
        gi = p[k]; gf = p[HID + k]; gg = p[2 * HID + k]; go = p[3 * HID + k];
    }

    for (int t = 0; t < T_STEPS; ++t) {
        // broadcast h across the wave: 20x v_readlane -> SGPRs
        float hb[HID];
#pragma unroll
        for (int j = 0; j < HID; ++j)
            hb[j] = __uint_as_float(
                (unsigned)__builtin_amdgcn_readlane((int)__float_as_uint(h), j));

        // prefetch next step's gate inputs while we compute this step
        float ni = 0.f, nf = 0.f, ng = 0.f, no_ = 0.f;
        if (act && t + 1 < T_STEPS) {
            const float* p = xg + (size_t)(t + 1) * 4 * HID;
            ni = p[k]; nf = p[HID + k]; ng = p[2 * HID + k]; no_ = p[3 * HID + k];
        }

        float ai = gi, af = gf, ag = gg, ao = go;
#pragma unroll
        for (int j = 0; j < HID; ++j) {
            ai = fmaf(Wi[j], hb[j], ai);
            af = fmaf(Wf[j], hb[j], af);
            ag = fmaf(Wg[j], hb[j], ag);
            ao = fmaf(Wo[j], hb[j], ao);
        }
        const float I = fast_sigmoid(ai);
        const float F = fast_sigmoid(af);
        const float G = fast_tanh(ag);
        const float O = fast_sigmoid(ao);
        c = fmaf(F, c, I * G);
        h = O * fast_tanh(c);

        if (act) lstm_buf[(size_t)t * HID + k] = h;

        gi = ni; gf = nf; gg = ng; go = no_;
    }
}

// ---------------------------------------------------------------------------
// Kernel 5: output heads.  One thread per timestep.
// out layout: mu[T*16] | log_std_out[T*16] | v[T] | log_std[T*16]
// ---------------------------------------------------------------------------
__global__ void heads_kernel(const float* __restrict__ lstm_buf,
                             const float* __restrict__ W_pfc, const float* __restrict__ b_pfc,
                             const float* __restrict__ W_mu,  const float* __restrict__ b_mu,
                             const float* __restrict__ W_ls,  const float* __restrict__ b_ls,
                             float* __restrict__ out) {
    const int t = blockIdx.x * blockDim.x + threadIdx.x;
    if (t >= T_STEPS) return;

    float hv[HID];
#pragma unroll
    for (int j = 0; j < HID; ++j) hv[j] = lstm_buf[(size_t)t * HID + j];

    float x[HID];
#pragma unroll 4
    for (int i = 0; i < HID; ++i) {
        float acc = b_pfc[i];
        const float* w = W_pfc + i * HID;
#pragma unroll
        for (int j = 0; j < HID; ++j) acc = fmaf(w[j], hv[j], acc);
        x[i] = fast_tanh(acc);
    }

    float* mu_out  = out;
    float* lso_out = out + (size_t)T_STEPS * NOUT;
    float* ls_out  = out + 2 * (size_t)T_STEPS * NOUT + T_STEPS;

#pragma unroll 4
    for (int o = 0; o < NOUT; ++o) {
        float am = b_mu[o], al = b_ls[o];
        const float* wm = W_mu + o * HID;
        const float* wl = W_ls + o * HID;
#pragma unroll
        for (int j = 0; j < HID; ++j) {
            am = fmaf(wm[j], x[j], am);
            al = fmaf(wl[j], x[j], al);
        }
        mu_out[(size_t)t * NOUT + o]  = fast_tanh(am);
        lso_out[(size_t)t * NOUT + o] = __expf(fmaxf(al, 0.0f) - LOG_STD_BOUND);
        ls_out[(size_t)t * NOUT + o]  = al;
    }
}

// ---------------------------------------------------------------------------
// Launch
// ---------------------------------------------------------------------------
extern "C" void kernel_launch(void* const* d_in, const int* in_sizes, int n_in,
                              void* d_out, int out_size, void* d_ws, size_t ws_size,
                              hipStream_t stream) {
    (void)in_sizes; (void)n_in; (void)out_size; (void)ws_size;

    const float* self_in = (const float*)d_in[0];
    const float* ally_in = (const float*)d_in[1];
    const int*   seg     = (const int*)  d_in[2];
    const float* W_self  = (const float*)d_in[3];
    const float* b_self  = (const float*)d_in[4];
    const float* W_ally  = (const float*)d_in[5];
    const float* b_ally  = (const float*)d_in[6];
    const float* W_cat   = (const float*)d_in[7];
    const float* b_cat   = (const float*)d_in[8];
    const float* W_ih    = (const float*)d_in[9];
    const float* W_hh    = (const float*)d_in[10];
    const float* b_ih    = (const float*)d_in[11];
    const float* b_hh    = (const float*)d_in[12];
    const float* W_pfc   = (const float*)d_in[13];
    const float* b_pfc   = (const float*)d_in[14];
    const float* W_vfc   = (const float*)d_in[15];
    const float* b_vfc   = (const float*)d_in[16];
    const float* W_mu    = (const float*)d_in[17];
    const float* b_mu    = (const float*)d_in[18];
    const float* W_ls    = (const float*)d_in[19];
    const float* b_ls    = (const float*)d_in[20];
    const float* W_v     = (const float*)d_in[21];
    const float* b_v     = (const float*)d_in[22];
    const float* h0      = (const float*)d_in[23];
    const float* c0      = (const float*)d_in[24];

    float* out = (float*)d_out;

    // workspace layout (floats)
    float* ws       = (float*)d_ws;
    float* sums     = ws;                               // T*HID
    float* cnt      = ws + (size_t)T_STEPS * HID;       // T
    float* sbuf     = cnt + T_STEPS;                    // T*HID
    float* xg       = sbuf + (size_t)T_STEPS * HID;     // T*4*HID
    float* lstm_buf = xg + (size_t)T_STEPS * 4 * HID;   // T*HID

    // 0) zero reduction buffers (sums + cnt = T*(HID+1) floats)
    {
        const int n = T_STEPS * (HID + 1);
        zero_kernel<<<(n + 255) / 256, 256, 0, stream>>>(sums, n);
    }

    // 1) ally MLP + segment reduction (HBM-bound; WMMA tiles)
    {
        const int nTiles = TOTAL_ROWS / 16;  // 65536
        ally_kernel<<<2048, 256, 0, stream>>>(ally_in, seg, W_ally, b_ally,
                                              sums, cnt, nTiles);
    }

    // 2) self MLP (WMMA tiles)
    {
        const int nTiles = T_STEPS / 16;     // 512
        self_kernel<<<64, 256, 0, stream>>>(self_in, W_self, b_self, sbuf, nTiles);
    }

    // 3) middle stage: avg / cat / gate inputs / value head
    mid_kernel<<<(T_STEPS + 255) / 256, 256, 0, stream>>>(
        sbuf, sums, cnt, W_cat, b_cat, W_ih, b_ih, b_hh,
        W_vfc, b_vfc, W_v, b_v, xg, out + 2 * (size_t)T_STEPS * NOUT);

    // 4) sequential LSTM (single wave32)
    lstm_kernel<<<1, 32, 0, stream>>>(xg, W_hh, h0, c0, lstm_buf);

    // 5) output heads
    heads_kernel<<<(T_STEPS + 255) / 256, 256, 0, stream>>>(
        lstm_buf, W_pfc, b_pfc, W_mu, b_mu, W_ls, b_ls, out);
}
